// Wiky_FusionWindowAttention_16844861735415
// MI455X (gfx1250) — compile-verified, matrix operand order checked by reading, not run on hardware
//
#include <hip/hip_runtime.h>
#include <stdint.h>

#define DIMC  180
#define NTOK  64
#define NHEAD 6
#define HD    30
#define KP    192      // DIM padded to 32-multiple
#define QKVN  540
#define QKVNP 544      // 34 tiles of 16
#define K2P   384      // 2*DIM padded
#define NWIN  4096

typedef __attribute__((ext_vector_type(16))) __bf16 v16bf;
typedef __attribute__((ext_vector_type(8)))  __bf16 v8bf;
typedef __attribute__((ext_vector_type(4)))  __bf16 v4bf;
typedef __attribute__((ext_vector_type(8)))  float  v8f;
typedef __attribute__((ext_vector_type(4)))  float  v4f;

static __device__ __forceinline__ v8f wmma_bf16(v16bf a, v16bf b, v8f c) {
  return __builtin_amdgcn_wmma_f32_16x16x32_bf16(false, a, false, b, (short)0, c,
                                                 false, false);
}

// Load one WMMA A/B operand for 16x16x32 bf16.
// rowbase = start of the 16-wide row this lane owns (m or n = lane%16).
// elems 0..7 -> K = k0 + 8*half + e ; elems 8..15 -> K = k0 + 16 + 8*half + (e-8)
static __device__ __forceinline__ v16bf load_op16(const __bf16* rowbase, int k0, int half) {
  const v8bf* p0 = (const v8bf*)(rowbase + k0 + half * 8);
  const v8bf* p1 = (const v8bf*)(rowbase + k0 + 16 + half * 8);
  v8bf lo = *p0, hi = *p1;
  v16bf r;
#pragma unroll
  for (int i = 0; i < 8; ++i) { r[i] = lo[i]; r[i + 8] = hi[i]; }
  return r;
}

static __device__ __forceinline__ v8f zero8() {
  v8f z = {0.f, 0.f, 0.f, 0.f, 0.f, 0.f, 0.f, 0.f};
  return z;
}

// Generic->LDS offset: LDS aperture maps generic addr[31:0] to the LDS byte offset.
static __device__ __forceinline__ unsigned lds_off(const void* p) {
  return (unsigned)(size_t)p;
}

// CDNA5 async global->LDS copy of 16 bytes (tracked by ASYNCcnt)
static __device__ __forceinline__ void async_copy_b128(unsigned ldsaddr, const void* gaddr) {
  asm volatile("global_load_async_to_lds_b128 %0, %1, off"
               :: "v"(ldsaddr), "v"((unsigned long long)(size_t)gaddr)
               : "memory");
}
static __device__ __forceinline__ void wait_async0() {
  asm volatile("s_wait_asynccnt 0" ::: "memory");
}

// ---------------------------------------------------------------- prep kernels
__global__ void bias_prep_kernel(const float* __restrict__ table,
                                 const int* __restrict__ rpi,
                                 float* __restrict__ bias_g) {
  int i = blockIdx.x * 256 + threadIdx.x;
  if (i >= NHEAD * NTOK * NTOK) return;
  int h = i / (NTOK * NTOK);
  int t = i % (NTOK * NTOK);
  bias_g[i] = table[rpi[t] * NHEAD + h];
}

// W (Kin x Nin, row-major f32) -> Wt (Npad x Kpad, bf16), zero padded
__global__ void wprep_kernel(const float* __restrict__ src, __bf16* __restrict__ dst,
                             int Kin, int Nin, int Kpad, int Npad) {
  int i = blockIdx.x * 256 + threadIdx.x;
  if (i >= Kpad * Npad) return;
  int n = i / Kpad, k = i % Kpad;
  float v = (n < Nin && k < Kin) ? src[k * Nin + n] : 0.f;
  dst[i] = (__bf16)v;
}

// Stage one 64x180 f32 window into LDS as bf16 64x192 (zero padded), vectorized.
static __device__ __forceinline__ void stage_x(const float* __restrict__ xg,
                                               __bf16* xl, int tid) {
  const int CH4 = DIMC / 4;  // 45 float4 per row
  for (int i = tid; i < NTOK * CH4; i += 256) {
    int r = i / CH4, c4 = (i % CH4) * 4;
    v4f v = *(const v4f*)(xg + r * DIMC + c4);
    v4bf o;
#pragma unroll
    for (int j = 0; j < 4; ++j) o[j] = (__bf16)v[j];
    *(v4bf*)(xl + r * KP + c4) = o;
  }
  for (int i = tid; i < NTOK * (KP - DIMC); i += 256) {
    int r = i / (KP - DIMC), c = DIMC + i % (KP - DIMC);
    xl[r * KP + c] = (__bf16)0.f;
  }
}

// ---------------------------------------------------------------- attention
__global__ __launch_bounds__(256)
void attn_kernel(const float* __restrict__ xe, const float* __restrict__ xb,
                 const float* __restrict__ maskg, const float* __restrict__ biasg,
                 const __bf16* __restrict__ wqkve_t, const __bf16* __restrict__ wqkvb_t,
                 const float* __restrict__ bqkv_e, const float* __restrict__ bqkv_b,
                 __bf16* __restrict__ out_e, __bf16* __restrict__ out_b) {
  __shared__ union {
    __bf16 x[NTOK * KP];                                          // phase 0/1
    struct { float s[NTOK * NTOK]; __bf16 p[NTOK * NTOK]; } sp;   // phase 2
  } u;
  __shared__ __bf16 q_lds[NTOK * KP];   // per-head 32-padded: col = h*32 + d
  __shared__ __bf16 k_lds[NTOK * KP];
  __shared__ __bf16 vt_lds[KP * NTOK];  // transposed V: row = h*32 + d
  __shared__ float  m_sh[NTOK * NTOK];  // mask window, shared across heads

  const int w  = blockIdx.x;
  const int br = blockIdx.y;
  const float*  xg   = (br ? xb : xe) + (size_t)w * NTOK * DIMC;
  const __bf16* Wt   = br ? wqkvb_t : wqkve_t;
  const float*  bq   = br ? bqkv_b : bqkv_e;
  __bf16*       outg = (br ? out_b : out_e) + (size_t)w * NTOK * DIMC;

  const int tid = threadIdx.x;
  const int wv = tid >> 5, l = tid & 31;
  const int lm = l & 15, lh = l >> 4;

  // async DMA of the mask window into LDS (overlaps the bf16 conversion below)
  for (int i = tid; i < NTOK * NTOK / 4; i += 256)
    async_copy_b128(lds_off(&m_sh[i * 4]), maskg + (size_t)w * NTOK * NTOK + i * 4);

  stage_x(xg, u.x, tid);
  // zero only the pad lanes of q/k/vt (d = 30,31 of each head)
  for (int i = tid; i < NTOK * 2 * NHEAD; i += 256) {
    int r = i / (2 * NHEAD), j = i % (2 * NHEAD);
    int col = (j >> 1) * 32 + 30 + (j & 1);
    q_lds[r * KP + col]  = (__bf16)0.f;
    k_lds[r * KP + col]  = (__bf16)0.f;
    vt_lds[col * NTOK + r] = (__bf16)0.f;
  }
  wait_async0();
  __syncthreads();

  // ---- phase 1: qkv = x @ Wqkv + b   (64x192 @ 192x544)
  // Each wave owns 2 m-tiles; wave pairs (same wv>>1) share an n-tile set -> Wt read twice per WG.
  const int m0 = (wv & 1) * 2;            // m-tiles m0, m0+1
  const int ncol = wv >> 1;               // 0..3
  const __bf16* arow0 = u.x + (m0 * 16 + lm) * KP;
  const __bf16* arow1 = u.x + ((m0 + 1) * 16 + lm) * KP;
  for (int g = 0; g < 9; ++g) {
    int n_tile = ncol + 4 * g;            // covers 0..33 across waves
    if (n_tile >= QKVNP / 16) break;
    const __bf16* brow = Wt + (n_tile * 16 + lm) * KP;
    v8f acc0 = zero8(), acc1 = zero8();
#pragma unroll
    for (int kb = 0; kb < 6; ++kb) {
      v16bf bm = load_op16(brow, kb * 32, lh);
      v16bf a0 = load_op16(arow0, kb * 32, lh);
      acc0 = wmma_bf16(a0, bm, acc0);
      v16bf a1 = load_op16(arow1, kb * 32, lh);
      acc1 = wmma_bf16(a1, bm, acc1);
    }
    int n = n_tile * 16 + lm;
    if (n < QKVN) {
      float badd = bq[n];
      int sec = n / DIMC, rr = n % DIMC;
      int hh = rr / HD, d = rr % HD;
      int col = hh * 32 + d;
#pragma unroll
      for (int r = 0; r < 8; ++r) {
        int ma = m0 * 16 + r + 8 * lh;
        int mb = ma + 16;
        float v0 = acc0[r] + badd, v1 = acc1[r] + badd;
        if (sec == 0)      { q_lds[ma * KP + col] = (__bf16)v0; q_lds[mb * KP + col] = (__bf16)v1; }
        else if (sec == 1) { k_lds[ma * KP + col] = (__bf16)v0; k_lds[mb * KP + col] = (__bf16)v1; }
        else               { vt_lds[col * NTOK + ma] = (__bf16)v0; vt_lds[col * NTOK + mb] = (__bf16)v1; }
      }
    }
  }
  __syncthreads();   // x dead; u.sp becomes live

  const float scale = 0.18257418583505536f;   // 30^-0.5
  for (int h = 0; h < NHEAD; ++h) {
    // ---- S = (q k^T)*scale + bias + mask : 16 tiles, K=32 (one wmma each)
#pragma unroll
    for (int t = wv; t < 16; t += 8) {
      int mi = t >> 2, nj = t & 3;
      v16bf a  = load_op16(q_lds + (mi * 16 + lm) * KP, h * 32, lh);
      v16bf bm = load_op16(k_lds + (nj * 16 + lm) * KP, h * 32, lh);
      v8f acc = wmma_bf16(a, bm, zero8());
      int n = nj * 16 + lm;
#pragma unroll
      for (int r = 0; r < 8; ++r) {
        int m = mi * 16 + r + 8 * lh;
        u.sp.s[m * NTOK + n] =
            acc[r] * scale + biasg[h * NTOK * NTOK + m * NTOK + n] + m_sh[m * NTOK + n];
      }
    }
    __syncthreads();
    // ---- softmax over rows (64 rows, one thread each)
    if (tid < NTOK) {
      float mx = -1e30f;
      for (int j = 0; j < NTOK; ++j) mx = fmaxf(mx, u.sp.s[tid * NTOK + j]);
      float sum = 0.f;
      for (int j = 0; j < NTOK; ++j) {
        float e = __expf(u.sp.s[tid * NTOK + j] - mx);
        u.sp.s[tid * NTOK + j] = e;
        sum += e;
      }
      float inv = 1.f / sum;
      for (int j = 0; j < NTOK; ++j)
        u.sp.p[tid * NTOK + j] = (__bf16)(u.sp.s[tid * NTOK + j] * inv);
    }
    __syncthreads();
    // ---- O = P @ V : 8 tiles (4m x 2n), one per wave, K=64 (two wmma)
    {
      int mi = wv & 3, nj = wv >> 2;
      v8f acc = zero8();
#pragma unroll
      for (int kb = 0; kb < 2; ++kb) {
        v16bf a  = load_op16(u.sp.p + (mi * 16 + lm) * NTOK, kb * 32, lh);
        v16bf bm = load_op16(vt_lds + (h * 32 + nj * 16 + lm) * NTOK, kb * 32, lh);
        acc = wmma_bf16(a, bm, acc);
      }
      int d = nj * 16 + lm;
      if (d < HD) {
#pragma unroll
        for (int r = 0; r < 8; ++r) {
          int m = mi * 16 + r + 8 * lh;
          outg[m * DIMC + h * HD + d] = (__bf16)acc[r];
        }
      }
    }
    __syncthreads();
  }
}

// ---------------------------------------------------------------- sigmoid gates
__global__ __launch_bounds__(256)
void sig_kernel(const float* __restrict__ xe, const float* __restrict__ xb,
                const __bf16* __restrict__ wt_e, const __bf16* __restrict__ wt_b,
                const float* __restrict__ bse, const float* __restrict__ bsb,
                __bf16* __restrict__ sige, __bf16* __restrict__ sigb) {
  __shared__ __bf16 x_lds[NTOK * KP];
  const int w  = blockIdx.x;
  const int br = blockIdx.y;
  const float*  xg  = (br ? xb : xe) + (size_t)w * NTOK * DIMC;
  const __bf16* Wt  = br ? wt_b : wt_e;
  const float*  bs  = br ? bsb : bse;
  __bf16*       out = (br ? sigb : sige) + (size_t)w * NTOK * DIMC;

  const int tid = threadIdx.x;
  const int wv = tid >> 5, l = tid & 31;
  const int lm = l & 15, lh = l >> 4;

  stage_x(xg, x_lds, tid);
  __syncthreads();

  const int m0 = (wv & 1) * 2;
  const __bf16* arow0 = x_lds + (m0 * 16 + lm) * KP;
  const __bf16* arow1 = x_lds + ((m0 + 1) * 16 + lm) * KP;
  for (int g = 0; g < 3; ++g) {
    int n_tile = (wv >> 1) + 4 * g;       // 0..11
    const __bf16* brow = Wt + (n_tile * 16 + lm) * KP;
    v8f acc0 = zero8(), acc1 = zero8();
#pragma unroll
    for (int kb = 0; kb < 6; ++kb) {
      v16bf bm = load_op16(brow, kb * 32, lh);
      v16bf a0 = load_op16(arow0, kb * 32, lh);
      acc0 = wmma_bf16(a0, bm, acc0);
      v16bf a1 = load_op16(arow1, kb * 32, lh);
      acc1 = wmma_bf16(a1, bm, acc1);
    }
    int n = n_tile * 16 + lm;
    if (n < DIMC) {
      float badd = bs[n];
#pragma unroll
      for (int r = 0; r < 8; ++r) {
        int ma = m0 * 16 + r + 8 * lh, mb = ma + 16;
        out[ma * DIMC + n] = (__bf16)(1.f / (1.f + __expf(-(acc0[r] + badd))));
        out[mb * DIMC + n] = (__bf16)(1.f / (1.f + __expf(-(acc1[r] + badd))));
      }
    }
  }
}

// ---------------------------------------------------------------- final projections
__global__ __launch_bounds__(256)
void final_kernel(const __bf16* __restrict__ out_e, const __bf16* __restrict__ out_b,
                  const __bf16* __restrict__ e_sig, const __bf16* __restrict__ b_sig,
                  const __bf16* __restrict__ wpe_t, const __bf16* __restrict__ wpb_t,
                  const float* __restrict__ bpe, const float* __restrict__ bpb,
                  float* __restrict__ dout) {
  __shared__ __bf16 a_lds[NTOK * K2P];
  const int w = blockIdx.x;
  const int y = blockIdx.y;  // 0 -> out_1 (gate e_sig, Wpe), 1 -> out_2 (gate b_sig, Wpb)
  const size_t base = (size_t)w * NTOK * DIMC;
  const __bf16* gate = (y ? b_sig : e_sig) + base;
  const __bf16* Wt   = y ? wpb_t : wpe_t;
  const float*  bo   = y ? bpb : bpe;
  float* outp = dout + (size_t)y * (size_t)NWIN * NTOK * DIMC + base;

  const int tid = threadIdx.x;
  const int wv = tid >> 5, l = tid & 31;
  const int lm = l & 15, lh = l >> 4;

  // gated concat rows: [out_e*g | out_b*g], bf16, zero padded to 384 (4-wide vectors)
  const int CH4 = DIMC / 4;  // 45
  for (int i = tid; i < NTOK * 2 * CH4; i += 256) {
    int m = i / (2 * CH4), j = i % (2 * CH4);
    const __bf16* src = (j < CH4) ? out_e : out_b;
    int c4 = (j % CH4) * 4;
    v4bf sv = *(const v4bf*)(src + base + m * DIMC + c4);
    v4bf gv = *(const v4bf*)(gate + m * DIMC + c4);
    v4bf o;
#pragma unroll
    for (int q = 0; q < 4; ++q) o[q] = (__bf16)((float)sv[q] * (float)gv[q]);
    int col = (j < CH4) ? c4 : DIMC + c4;
    *(v4bf*)(a_lds + m * K2P + col) = o;
  }
  for (int i = tid; i < NTOK * (K2P - 2 * DIMC); i += 256) {
    int m = i / (K2P - 2 * DIMC), c = 2 * DIMC + i % (K2P - 2 * DIMC);
    a_lds[m * K2P + c] = (__bf16)0.f;
  }
  __syncthreads();

  const int m0 = (wv & 1) * 2;
  const __bf16* arow0 = a_lds + (m0 * 16 + lm) * K2P;
  const __bf16* arow1 = a_lds + ((m0 + 1) * 16 + lm) * K2P;
  for (int g = 0; g < 3; ++g) {
    int n_tile = (wv >> 1) + 4 * g;       // 0..11
    const __bf16* brow = Wt + (n_tile * 16 + lm) * K2P;
    v8f acc0 = zero8(), acc1 = zero8();
#pragma unroll
    for (int kb = 0; kb < 12; ++kb) {
      v16bf bm = load_op16(brow, kb * 32, lh);
      v16bf a0 = load_op16(arow0, kb * 32, lh);
      acc0 = wmma_bf16(a0, bm, acc0);
      v16bf a1 = load_op16(arow1, kb * 32, lh);
      acc1 = wmma_bf16(a1, bm, acc1);
    }
    int n = n_tile * 16 + lm;
    if (n < DIMC) {
      float badd = bo[n];
#pragma unroll
      for (int r = 0; r < 8; ++r) {
        int ma = m0 * 16 + r + 8 * lh, mb = ma + 16;
        outp[ma * DIMC + n] = acc0[r] + badd;
        outp[mb * DIMC + n] = acc1[r] + badd;
      }
    }
  }
}

// ---------------------------------------------------------------- launch
extern "C" void kernel_launch(void* const* d_in, const int* in_sizes, int n_in,
                              void* d_out, int out_size, void* d_ws, size_t ws_size,
                              hipStream_t stream) {
  const float* e       = (const float*)d_in[0];
  const float* b       = (const float*)d_in[1];
  const float* mask    = (const float*)d_in[2];
  const float* rpb     = (const float*)d_in[3];
  const int*   rpi     = (const int*)d_in[4];
  const float* Wqkv_e  = (const float*)d_in[5];
  const float* bqkv_e  = (const float*)d_in[6];
  const float* Wqkv_b  = (const float*)d_in[7];
  const float* bqkv_b  = (const float*)d_in[8];
  const float* Wpe_s   = (const float*)d_in[9];
  const float* bpe_s   = (const float*)d_in[10];
  const float* Wpb_s   = (const float*)d_in[11];
  const float* bpb_s   = (const float*)d_in[12];
  const float* Wpe     = (const float*)d_in[13];
  const float* bpe     = (const float*)d_in[14];
  const float* Wpb     = (const float*)d_in[15];
  const float* bpb     = (const float*)d_in[16];
  (void)in_sizes; (void)n_in; (void)out_size; (void)ws_size;

  // workspace layout (16B-aligned blocks)
  char* ws = (char*)d_ws;
  float*  bias_g  = (float*)ws;                       // 6*64*64 f32 = 98304 B
  __bf16* wqkve_t = (__bf16*)(ws + 98304);            // 544*192
  __bf16* wqkvb_t = wqkve_t + 544 * 192;
  __bf16* wpes_t  = wqkvb_t + 544 * 192;              // 192*192
  __bf16* wpbs_t  = wpes_t + 192 * 192;
  __bf16* wpe_t   = wpbs_t + 192 * 192;               // 192*384
  __bf16* wpb_t   = wpe_t + 192 * 384;
  __bf16* out_e   = wpb_t + 192 * 384;                // 4096*64*180
  __bf16* out_b   = out_e + (size_t)NWIN * NTOK * DIMC;
  __bf16* e_sig   = out_b + (size_t)NWIN * NTOK * DIMC;
  __bf16* b_sig   = e_sig + (size_t)NWIN * NTOK * DIMC;

  bias_prep_kernel<<<(NHEAD * NTOK * NTOK + 255) / 256, 256, 0, stream>>>(rpb, rpi, bias_g);
  wprep_kernel<<<(544 * 192 + 255) / 256, 256, 0, stream>>>(Wqkv_e, wqkve_t, DIMC, QKVN, KP, QKVNP);
  wprep_kernel<<<(544 * 192 + 255) / 256, 256, 0, stream>>>(Wqkv_b, wqkvb_t, DIMC, QKVN, KP, QKVNP);
  wprep_kernel<<<(192 * 192 + 255) / 256, 256, 0, stream>>>(Wpe_s, wpes_t, DIMC, DIMC, KP, 192);
  wprep_kernel<<<(192 * 192 + 255) / 256, 256, 0, stream>>>(Wpb_s, wpbs_t, DIMC, DIMC, KP, 192);
  wprep_kernel<<<(192 * 384 + 255) / 256, 256, 0, stream>>>(Wpe, wpe_t, 2 * DIMC, DIMC, K2P, 192);
  wprep_kernel<<<(192 * 384 + 255) / 256, 256, 0, stream>>>(Wpb, wpb_t, 2 * DIMC, DIMC, K2P, 192);

  dim3 g2(NWIN, 2);
  attn_kernel<<<g2, 256, 0, stream>>>(e, b, mask, bias_g, wqkve_t, wqkvb_t,
                                      bqkv_e, bqkv_b, out_e, out_b);
  sig_kernel<<<g2, 256, 0, stream>>>(e, b, wpes_t, wpbs_t, bpe_s, bpb_s, e_sig, b_sig);
  final_kernel<<<g2, 256, 0, stream>>>(out_e, out_b, e_sig, b_sig, wpe_t, wpb_t,
                                       bpe, bpb, (float*)d_out);
}
